// normal_estimate_net_54820962566385
// MI455X (gfx1250) — matrix operand order repaired and checked by reading, not compiled, not run on hardware
//
#include <hip/hip_runtime.h>
#include <hip/hip_bf16.h>

#define Bb 8
#define Nn 4096

typedef __attribute__((ext_vector_type(16))) _Float16 v16h;
typedef __attribute__((ext_vector_type(8)))  _Float16 v8h;
typedef __attribute__((ext_vector_type(8)))  float    v8f;

union hfrag {
    v16h v;
    v8h  h2[2];
};

__device__ __forceinline__ v8f wmma_f16(v16h a, v16h b, v8f c) {
    return __builtin_amdgcn_wmma_f32_16x16x32_f16(false, a, false, b, (short)0, c, false, false);
}

// Build A fragment from a row-major [M][K] panel row: two contiguous 8-half runs
// at K = kc + 8h and K = kc + 16 + 8h (ISA 16-bit A-matrix 16x32 layout).
__device__ __forceinline__ v16h load_afrag(const _Float16* rowPtr, int kc, int h) {
    hfrag f;
    f.h2[0] = *(const v8h*)(rowPtr + kc + 8 * h);
    f.h2[1] = *(const v8h*)(rowPtr + kc + 16 + 8 * h);
    return f.v;
}
// Build B fragment from a col-major [N][K] panel row: two contiguous 8-half runs
// at K = kc + 8h and K = kc + 16 + 8h (same K order as the A fragment).
__device__ __forceinline__ v16h load_bfrag(const _Float16* colPtr, int kc, int h) {
    hfrag f;
    f.h2[0] = *(const v8h*)(colPtr + kc + 8 * h);
    f.h2[1] = *(const v8h*)(colPtr + kc + 16 + 8 * h);
    return f.v;
}

// ---------------------------------------------------------------------------
// K1: exact kNN (top-32, sorted desc) on raw point cloud (C=3)
// grid (Nn, Bb), block 256
// ---------------------------------------------------------------------------
__global__ void __launch_bounds__(256) ne_knn_pc(const float* __restrict__ pc,
                                                 int* __restrict__ idxp) {
    __shared__ float d[Nn];
    __shared__ float rv[256];
    __shared__ int   ri[256];
    const int tid = threadIdx.x;
    const int i = blockIdx.x, b = blockIdx.y;
    const float xi0 = pc[((size_t)b * 3 + 0) * Nn + i];
    const float xi1 = pc[((size_t)b * 3 + 1) * Nn + i];
    const float xi2 = pc[((size_t)b * 3 + 2) * Nn + i];
    for (int j = tid; j < Nn; j += 256) {
        float d0 = pc[((size_t)b * 3 + 0) * Nn + j] - xi0;
        float d1 = pc[((size_t)b * 3 + 1) * Nn + j] - xi1;
        float d2 = pc[((size_t)b * 3 + 2) * Nn + j] - xi2;
        d[j] = -(d0 * d0 + d1 * d1 + d2 * d2);
    }
    __syncthreads();
    for (int r = 0; r < 32; ++r) {
        float bv = -3.4e38f; int bi = Nn;
        for (int j = tid; j < Nn; j += 256) {
            float v = d[j];
            if (v > bv) { bv = v; bi = j; }
        }
        rv[tid] = bv; ri[tid] = bi;
        __syncthreads();
        for (int s = 128; s > 0; s >>= 1) {
            if (tid < s) {
                if (rv[tid + s] > rv[tid] ||
                    (rv[tid + s] == rv[tid] && ri[tid + s] < ri[tid])) {
                    rv[tid] = rv[tid + s]; ri[tid] = ri[tid + s];
                }
            }
            __syncthreads();
        }
        if (tid == 0) {
            idxp[((size_t)b * Nn + i) * 32 + r] = ri[0];
            d[ri[0]] = -3.4e38f;
        }
        __syncthreads();
    }
}

// ---------------------------------------------------------------------------
// K2: small edge block (Cin=6), pass1: GroupNorm(2) sum/sumsq
// grid (Nn*k/256, Bb), block 256
// ---------------------------------------------------------------------------
__global__ void __launch_bounds__(256) ne_small_conv_pass1(
        const float* __restrict__ pc, const int* __restrict__ idx,
        const float* __restrict__ W, float* __restrict__ stats, int k) {
    __shared__ float sW[64 * 6];
    __shared__ float sred[4];
    const int tid = threadIdx.x;
    for (int t = tid; t < 384; t += 256) sW[t] = W[t];
    if (tid < 4) sred[tid] = 0.f;
    __syncthreads();
    const int b = blockIdx.y;
    const int e = blockIdx.x * 256 + tid;
    const int kk = e % k;
    const int n  = e / k;
    const int j  = idx[((size_t)b * Nn + n) * 32 + kk];
    float F[6];
#pragma unroll
    for (int c = 0; c < 3; ++c) {
        float ctr = pc[((size_t)b * 3 + c) * Nn + n];
        float nb  = pc[((size_t)b * 3 + c) * Nn + j];
        F[c] = nb - ctr; F[c + 3] = ctr;
    }
    float s0 = 0.f, s1 = 0.f, q0 = 0.f, q1 = 0.f;
    for (int o = 0; o < 64; ++o) {
        float y = 0.f;
#pragma unroll
        for (int i2 = 0; i2 < 6; ++i2) y += sW[o * 6 + i2] * F[i2];
        if (o < 32) { s0 += y; q0 += y * y; } else { s1 += y; q1 += y * y; }
    }
    atomicAdd(&sred[0], s0); atomicAdd(&sred[1], q0);
    atomicAdd(&sred[2], s1); atomicAdd(&sred[3], q1);
    __syncthreads();
    if (tid < 4)
        atomicAdd(&stats[((size_t)b * 2 + (tid >> 1)) * 2 + (tid & 1)], sred[tid]);
}

// ---------------------------------------------------------------------------
// K3: small edge block pass2: GN + LeakyReLU(0.1) + max over k (+ addend)
// grid (Nn, Bb), block 64
// ---------------------------------------------------------------------------
__global__ void __launch_bounds__(64) ne_small_conv_pass2(
        const float* __restrict__ pc, const int* __restrict__ idx,
        const float* __restrict__ W, const float* __restrict__ gw,
        const float* __restrict__ gb, const float* __restrict__ stats,
        float inv_cnt, const float* __restrict__ addend,
        float* __restrict__ xout, int k) {
    __shared__ int jl[32];
    const int o = threadIdx.x;
    const int n = blockIdx.x, b = blockIdx.y;
    if (o < k) jl[o] = idx[((size_t)b * Nn + n) * 32 + o];
    __syncthreads();
    const int g = o >> 5;
    const float mean = stats[((size_t)b * 2 + g) * 2 + 0] * inv_cnt;
    const float var  = stats[((size_t)b * 2 + g) * 2 + 1] * inv_cnt - mean * mean;
    const float rstd = rsqrtf(var + 1e-5f);
    float w6[6];
#pragma unroll
    for (int i2 = 0; i2 < 6; ++i2) w6[i2] = W[o * 6 + i2];
    const float c0 = pc[((size_t)b * 3 + 0) * Nn + n];
    const float c1 = pc[((size_t)b * 3 + 1) * Nn + n];
    const float c2 = pc[((size_t)b * 3 + 2) * Nn + n];
    const float gwv = gw[o], gbv = gb[o];
    float mx = -3.4e38f;
    for (int kk = 0; kk < k; ++kk) {
        int j = jl[kk];
        float d0 = pc[((size_t)b * 3 + 0) * Nn + j] - c0;
        float d1 = pc[((size_t)b * 3 + 1) * Nn + j] - c1;
        float d2 = pc[((size_t)b * 3 + 2) * Nn + j] - c2;
        float y = w6[0] * d0 + w6[1] * d1 + w6[2] * d2 +
                  w6[3] * c0 + w6[4] * c1 + w6[5] * c2;
        y = (y - mean) * rstd * gwv + gbv;
        y = (y >= 0.f) ? y : 0.1f * y;
        mx = fmaxf(mx, y);
    }
    if (addend) mx += addend[((size_t)b * 64 + o) * Nn + n];
    xout[((size_t)b * 64 + o) * Nn + n] = mx;
}

// ---------------------------------------------------------------------------
// cvt f32 -> f16
// ---------------------------------------------------------------------------
__global__ void ne_cvt(const float* __restrict__ x, _Float16* __restrict__ xh, int n) {
    int gi = blockIdx.x * 256 + threadIdx.x;
    if (gi < n) xh[gi] = (_Float16)x[gi];
}

// ---------------------------------------------------------------------------
// squared norms of 64-ch half features: xx[b*N+n]
// ---------------------------------------------------------------------------
__global__ void __launch_bounds__(256) ne_sqnorm(const _Float16* __restrict__ xh,
                                                 float* __restrict__ xx) {
    int gi = blockIdx.x * 256 + threadIdx.x;   // Bb*Nn total
    int b = gi >> 12, n = gi & (Nn - 1);
    float s = 0.f;
    for (int c = 0; c < 64; ++c) {
        float v = (float)xh[((size_t)b * 64 + c) * Nn + n];
        s += v * v;
    }
    xx[gi] = s;
}

// ---------------------------------------------------------------------------
// K7: feature kNN via WMMA inner-product strips + fused sorted top-32
// grid (Nn/16, Bb), block 256 (8 waves)
// ---------------------------------------------------------------------------
__global__ void __launch_bounds__(256) ne_knn_feat(const _Float16* __restrict__ xh,
                                                   const float* __restrict__ xx,
                                                   int* __restrict__ idxf) {
    __shared__ _Float16 sA[16][72];     // [row][K]   (A: row-major, K-contig)
    __shared__ _Float16 sBt[128][72];   // [col][K]   (B: col-major, K-contig)
    __shared__ float    sPD[16][128];
    __shared__ float    tv[16][32];
    __shared__ int      ti[16][32];
    __shared__ float    sxx[16];
    const int tid = threadIdx.x;
    const int lane = tid & 31, w = tid >> 5;
    const int i0 = blockIdx.x * 16, b = blockIdx.y;
    for (int t = tid; t < 16 * 64; t += 256) {
        int m = t >> 6, ch = t & 63;
        sA[m][ch] = xh[((size_t)b * 64 + ch) * Nn + i0 + m];
    }
    for (int t = tid; t < 512; t += 256) tv[t >> 5][t & 31] = -3.4e38f;
    if (tid < 16) sxx[tid] = xx[(size_t)b * Nn + i0 + tid];
    __syncthreads();
    const int h = lane >> 4, nl = lane & 15;
    for (int cb = 0; cb < Nn / 128; ++cb) {
        const int c0 = cb * 128;
        for (int t = tid; t < 64 * 128; t += 256) {
            int ch = t >> 7, cc = t & 127;       // coalesced global reads along cc
            sBt[cc][ch] = xh[((size_t)b * 64 + ch) * Nn + c0 + cc];
        }
        if (cb + 1 < Nn / 128)
            __builtin_prefetch(&xh[((size_t)b * 64 + (tid >> 7)) * Nn + c0 + 128], 0, 3);
        __syncthreads();
        {
            const int j0 = w * 16;
            const _Float16* aRow = &sA[nl][0];
            const _Float16* bCol = &sBt[j0 + nl][0];
            v8f acc;
#pragma unroll
            for (int r = 0; r < 8; ++r) acc[r] = 0.f;
#pragma unroll
            for (int kc = 0; kc < 64; kc += 32) {
                v16h af = load_afrag(aRow, kc, h);
                v16h bf = load_bfrag(bCol, kc, h);
                acc = wmma_f16(af, bf, acc);
            }
#pragma unroll
            for (int r = 0; r < 8; ++r) {
                int m = r + 8 * h;
                int jj = c0 + j0 + nl;
                sPD[m][j0 + nl] = 2.f * acc[r] - sxx[m] - xx[(size_t)b * Nn + jj];
            }
        }
        __syncthreads();
        if (tid < 16) {
            const int row = tid;
            for (int c = 0; c < 128; ++c) {
                float v = sPD[row][c];
                if (v > tv[row][31]) {
                    int p = 31;
                    while (p > 0 && tv[row][p - 1] < v) {
                        tv[row][p] = tv[row][p - 1];
                        ti[row][p] = ti[row][p - 1];
                        --p;
                    }
                    tv[row][p] = v; ti[row][p] = c0 + c;
                }
            }
        }
        __syncthreads();
    }
    for (int t = tid; t < 512; t += 256)
        idxf[((size_t)b * Nn + i0 + (t >> 5)) * 32 + (t & 31)] = ti[t >> 5][t & 31];
}

// ---------------------------------------------------------------------------
// K8/K9: big edge block (Cin=128) via WMMA. pass=1 GN sums, pass=2 output.
// grid (Nn/16, Bb), block 256 (8 waves)
// ---------------------------------------------------------------------------
__global__ void __launch_bounds__(256) ne_big_conv(
        const _Float16* __restrict__ xh, const int* __restrict__ idx,
        const float* __restrict__ W, const float* __restrict__ gw,
        const float* __restrict__ gb, float* __restrict__ stats,
        float inv_cnt, float* __restrict__ xout, int pass) {
    __shared__ _Float16 sW[64][136];        // [o][K]  A panel, K-contig
    __shared__ _Float16 sXc[64][18];
    __shared__ int      sJ[16][32];
    __shared__ _Float16 sFt[8][16][136];    // per-wave B panel, [col][K] K-contig
    __shared__ float    sred[4];
    const int tid = threadIdx.x, lane = tid & 31, w = tid >> 5;
    const int n0 = blockIdx.x * 16, b = blockIdx.y;
    for (int t = tid; t < 64 * 128; t += 256) {
        int o = t >> 7, ch = t & 127;
        sW[o][ch] = (_Float16)W[o * 128 + ch];
    }
    for (int t = tid; t < 64 * 16; t += 256) {
        int ch = t >> 4, nn = t & 15;
        sXc[ch][nn] = xh[((size_t)b * 64 + ch) * Nn + n0 + nn];
    }
    for (int t = tid; t < 512; t += 256)
        sJ[t >> 5][t & 31] = idx[((size_t)b * Nn + n0 + (t >> 5)) * 32 + (t & 31)];
    if (tid < 4) sred[tid] = 0.f;
    __syncthreads();
    const int h = lane >> 4, nl = lane & 15;
    float mean[2], rstd[2];
    if (pass == 2) {
#pragma unroll
        for (int g = 0; g < 2; ++g) {
            mean[g] = stats[((size_t)b * 2 + g) * 2 + 0] * inv_cnt;
            float var = stats[((size_t)b * 2 + g) * 2 + 1] * inv_cnt - mean[g] * mean[g];
            rstd[g] = rsqrtf(var + 1e-5f);
        }
    }
    float ls[2] = {0.f, 0.f}, lq[2] = {0.f, 0.f};
    for (int np = 0; np < 2; ++np) {
        const int nloc = 2 * w + np;
        v8f rmax[4];
        if (pass == 2) {
#pragma unroll
            for (int rt = 0; rt < 4; ++rt)
#pragma unroll
                for (int r = 0; r < 8; ++r) rmax[rt][r] = -3.4e38f;
        }
        for (int ct = 0; ct < 2; ++ct) {
            // build this wave's gathered edge-feature panel, [col][K]
            for (int t = lane; t < 128 * 16; t += 32) {
                int ch = t >> 4, cc = t & 15;
                int j = sJ[nloc][ct * 16 + cc];
                _Float16 v;
                if (ch < 64) v = xh[((size_t)b * 64 + ch) * Nn + j] - sXc[ch][nloc];
                else         v = sXc[ch - 64][nloc];
                sFt[w][cc][ch] = v;
            }
            const _Float16* bCol = &sFt[w][nl][0];
            v8f acc[4];
#pragma unroll
            for (int rt = 0; rt < 4; ++rt)
#pragma unroll
                for (int r = 0; r < 8; ++r) acc[rt][r] = 0.f;
#pragma unroll
            for (int kc = 0; kc < 128; kc += 32) {
                v16h bf = load_bfrag(bCol, kc, h);
#pragma unroll
                for (int rt = 0; rt < 4; ++rt) {
                    v16h af = load_afrag(&sW[16 * rt + nl][0], kc, h);
                    acc[rt] = wmma_f16(af, bf, acc[rt]);
                }
            }
            if (pass == 1) {
#pragma unroll
                for (int rt = 0; rt < 4; ++rt) {
                    int g = rt >> 1;
#pragma unroll
                    for (int r = 0; r < 8; ++r) {
                        float y = acc[rt][r];
                        ls[g] += y; lq[g] += y * y;
                    }
                }
            } else {
#pragma unroll
                for (int rt = 0; rt < 4; ++rt) {
#pragma unroll
                    for (int r = 0; r < 8; ++r) {
                        int m = 16 * rt + r + 8 * h;
                        int g = m >> 5;
                        float y = (acc[rt][r] - mean[g]) * rstd[g] * gw[m] + gb[m];
                        y = (y >= 0.f) ? y : 0.1f * y;
                        rmax[rt][r] = fmaxf(rmax[rt][r], y);
                    }
                }
            }
        }
        if (pass == 2) {
#pragma unroll
            for (int rt = 0; rt < 4; ++rt) {
#pragma unroll
                for (int r = 0; r < 8; ++r) {
                    float v = rmax[rt][r];
#pragma unroll
                    for (int mm = 1; mm < 16; mm <<= 1)
                        v = fmaxf(v, __shfl_xor(v, mm, 32));
                    if (nl == 0)
                        xout[((size_t)b * 64 + 16 * rt + r + 8 * h) * Nn + n0 + nloc] = v;
                }
            }
        }
    }
    if (pass == 1) {
        atomicAdd(&sred[0], ls[0]); atomicAdd(&sred[1], lq[0]);
        atomicAdd(&sred[2], ls[1]); atomicAdd(&sred[3], lq[1]);
        __syncthreads();
        if (tid < 4)
            atomicAdd(&stats[((size_t)b * 2 + (tid >> 1)) * 2 + (tid & 1)], sred[tid]);
    }
}

// ---------------------------------------------------------------------------
// K14: fm layer GEMM (256x192 x cat[192,N]) + bias + GN4 sums
// grid (Nn/64, Bb), block 256 (8 waves)
// ---------------------------------------------------------------------------
__global__ void __launch_bounds__(256) ne_fm(
        const float* __restrict__ x1, const float* __restrict__ x2,
        const float* __restrict__ x3, const _Float16* __restrict__ fmwh,
        const float* __restrict__ fmb, float* __restrict__ yfm,
        float* __restrict__ stats4) {
    __shared__ _Float16 sBt[64][200];   // [col][K] B panel, K-contig
    __shared__ float sred[8];
    const int tid = threadIdx.x, lane = tid & 31, w = tid >> 5;
    const int n0 = blockIdx.x * 64, b = blockIdx.y;
    for (int t = tid; t < 192 * 64; t += 256) {
        int ch = t >> 6, cc = t & 63;   // coalesced global reads along cc
        const float* src = (ch < 64)  ? (x1 + ((size_t)b * 64 + ch) * Nn)
                         : (ch < 128) ? (x2 + ((size_t)b * 64 + ch - 64) * Nn)
                                      : (x3 + ((size_t)b * 64 + ch - 128) * Nn);
        sBt[cc][ch] = (_Float16)src[n0 + cc];
    }
    if (tid < 8) sred[tid] = 0.f;
    __syncthreads();
    const int h = lane >> 4, nl = lane & 15;
    float ls[4] = {0.f, 0.f, 0.f, 0.f}, lq[4] = {0.f, 0.f, 0.f, 0.f};
    for (int ct = 0; ct < 4; ++ct) {
        const _Float16* bCol = &sBt[ct * 16 + nl][0];
        const _Float16* aRow0 = fmwh + (size_t)(16 * (2 * w) + nl) * 192;
        const _Float16* aRow1 = fmwh + (size_t)(16 * (2 * w + 1) + nl) * 192;
        v8f acc0, acc1;
#pragma unroll
        for (int r = 0; r < 8; ++r) { acc0[r] = 0.f; acc1[r] = 0.f; }
#pragma unroll
        for (int kc = 0; kc < 192; kc += 32) {
            v16h bf  = load_bfrag(bCol, kc, h);
            v16h af0 = load_afrag(aRow0, kc, h);
            v16h af1 = load_afrag(aRow1, kc, h);
            acc0 = wmma_f16(af0, bf, acc0);
            acc1 = wmma_f16(af1, bf, acc1);
        }
        const int n = n0 + ct * 16 + nl;
#pragma unroll
        for (int r = 0; r < 8; ++r) {
            int o = 16 * (2 * w) + r + 8 * h;
            float y = acc0[r] + fmb[o];
            yfm[((size_t)b * 256 + o) * Nn + n] = y;
            int g = o >> 6; ls[g] += y; lq[g] += y * y;
            int o2 = 16 * (2 * w + 1) + r + 8 * h;
            float y2 = acc1[r] + fmb[o2];
            yfm[((size_t)b * 256 + o2) * Nn + n] = y2;
            int g2 = o2 >> 6; ls[g2] += y2; lq[g2] += y2 * y2;
        }
    }
#pragma unroll
    for (int g = 0; g < 4; ++g) {
        atomicAdd(&sred[g * 2 + 0], ls[g]);
        atomicAdd(&sred[g * 2 + 1], lq[g]);
    }
    __syncthreads();
    if (tid < 8)
        atomicAdd(&stats4[((size_t)b * 4 + (tid >> 1)) * 2 + (tid & 1)], sred[tid]);
}

// ---------------------------------------------------------------------------
// K15: head = GN4 + LeakyReLU(0.01) + 3x256 matmul + bias + L2 normalize
// ---------------------------------------------------------------------------
__global__ void __launch_bounds__(256) ne_head(
        const float* __restrict__ yfm, const float* __restrict__ stats4,
        const float* __restrict__ gn4w, const float* __restrict__ gn4b,
        const float* __restrict__ hw, const float* __restrict__ hb,
        float* __restrict__ out) {
    const int gi = blockIdx.x * 256 + threadIdx.x;   // Bb*Nn
    const int b = gi >> 12, n = gi & (Nn - 1);
    const float inv = 1.f / (64.f * 4096.f);
    float mean[4], rstd[4];
#pragma unroll
    for (int g = 0; g < 4; ++g) {
        mean[g] = stats4[((size_t)b * 4 + g) * 2 + 0] * inv;
        float var = stats4[((size_t)b * 4 + g) * 2 + 1] * inv - mean[g] * mean[g];
        rstd[g] = rsqrtf(var + 1e-5f);
    }
    float p0 = 0.f, p1 = 0.f, p2 = 0.f;
    for (int o = 0; o < 256; ++o) {
        float v = yfm[((size_t)b * 256 + o) * Nn + n];
        int g = o >> 6;
        float t = (v - mean[g]) * rstd[g] * gn4w[o] + gn4b[o];
        t = (t >= 0.f) ? t : 0.01f * t;
        p0 += hw[0 * 256 + o] * t;
        p1 += hw[1 * 256 + o] * t;
        p2 += hw[2 * 256 + o] * t;
    }
    p0 += hb[0]; p1 += hb[1]; p2 += hb[2];
    float nrm = sqrtf(p0 * p0 + p1 * p1 + p2 * p2);
    float s = 1.f / fmaxf(nrm, 1e-12f);
    out[(size_t)gi * 3 + 0] = p0 * s;
    out[(size_t)gi * 3 + 1] = p1 * s;
    out[(size_t)gi * 3 + 2] = p2 * s;
}

// ---------------------------------------------------------------------------
extern "C" void kernel_launch(void* const* d_in, const int* in_sizes, int n_in,
                              void* d_out, int out_size, void* d_ws, size_t ws_size,
                              hipStream_t stream) {
    (void)in_sizes; (void)n_in; (void)out_size; (void)ws_size;
    const float* pc      = (const float*)d_in[0];
    const float* conv0_w = (const float*)d_in[1];
    const float* gn0_w   = (const float*)d_in[2];
    const float* gn0_b   = (const float*)d_in[3];
    const float* conv1_w = (const float*)d_in[4];
    const float* gn1_w   = (const float*)d_in[5];
    const float* gn1_b   = (const float*)d_in[6];
    const float* conv2_w = (const float*)d_in[7];
    const float* gn2_w   = (const float*)d_in[8];
    const float* gn2_b   = (const float*)d_in[9];
    const float* conv3_w = (const float*)d_in[10];
    const float* gn3_w   = (const float*)d_in[11];
    const float* gn3_b   = (const float*)d_in[12];
    const float* fm_w    = (const float*)d_in[13];
    const float* fm_b    = (const float*)d_in[14];
    const float* gn4_w   = (const float*)d_in[15];
    const float* gn4_b   = (const float*)d_in[16];
    const float* head_w  = (const float*)d_in[17];
    const float* head_b  = (const float*)d_in[18];

    char* ws = (char*)d_ws;
    const size_t BN   = (size_t)Bb * Nn;
    const size_t B64N = (size_t)Bb * 64 * Nn;
    int*      idxp = (int*)ws;            ws += BN * 32 * sizeof(int);
    int*      idxf = (int*)ws;            ws += BN * 32 * sizeof(int);
    float*    x0f  = (float*)ws;          ws += B64N * sizeof(float);
    float*    x1f  = (float*)ws;          ws += B64N * sizeof(float);
    float*    x2f  = (float*)ws;          ws += B64N * sizeof(float);
    float*    x3f  = (float*)ws;          ws += B64N * sizeof(float);
    _Float16* xh   = (_Float16*)ws;       ws += B64N * sizeof(_Float16);
    float*    xx   = (float*)ws;          ws += BN * sizeof(float);
    _Float16* fmwh = (_Float16*)ws;       ws += (size_t)256 * 192 * sizeof(_Float16);
    float*    yfm  = (float*)ws;          ws += (size_t)Bb * 256 * Nn * sizeof(float);
    float*    gacc = (float*)ws;          // 192 floats of GN accumulators

    hipMemsetAsync(gacc, 0, 192 * sizeof(float), stream);
    float* s0 = gacc;       // conv0: [B][2][2]
    float* s1 = gacc + 32;  // conv1
    float* s2 = gacc + 64;  // conv2
    float* s3 = gacc + 96;  // conv3
    float* s4 = gacc + 128; // fm: [B][4][2]

    const float inv_c0 = 1.f / (32.f * Nn * 16.f);
    const float inv_ck = 1.f / (32.f * Nn * 32.f);

    // ---- kNN on raw points (shared by edge blocks 0 & 1) ----
    ne_knn_pc<<<dim3(Nn, Bb), 256, 0, stream>>>(pc, idxp);

    // ---- edge block 0 (k=16) -> x0 ----
    ne_small_conv_pass1<<<dim3(Nn * 16 / 256, Bb), 256, 0, stream>>>(pc, idxp, conv0_w, s0, 16);
    ne_small_conv_pass2<<<dim3(Nn, Bb), 64, 0, stream>>>(pc, idxp, conv0_w, gn0_w, gn0_b,
                                                         s0, inv_c0, nullptr, x0f, 16);
    // ---- edge block 1 (k=32) -> x1 = max + x0 ----
    ne_small_conv_pass1<<<dim3(Nn * 32 / 256, Bb), 256, 0, stream>>>(pc, idxp, conv1_w, s1, 32);
    ne_small_conv_pass2<<<dim3(Nn, Bb), 64, 0, stream>>>(pc, idxp, conv1_w, gn1_w, gn1_b,
                                                         s1, inv_ck, x0f, x1f, 32);

    // ---- kNN(x1) + edge block 2 -> x2 ----
    ne_cvt<<<(int)((B64N + 255) / 256), 256, 0, stream>>>(x1f, xh, (int)B64N);
    ne_sqnorm<<<(int)(BN / 256), 256, 0, stream>>>(xh, xx);
    ne_knn_feat<<<dim3(Nn / 16, Bb), 256, 0, stream>>>(xh, xx, idxf);
    ne_big_conv<<<dim3(Nn / 16, Bb), 256, 0, stream>>>(xh, idxf, conv2_w, gn2_w, gn2_b,
                                                       s2, inv_ck, x2f, 1);
    ne_big_conv<<<dim3(Nn / 16, Bb), 256, 0, stream>>>(xh, idxf, conv2_w, gn2_w, gn2_b,
                                                       s2, inv_ck, x2f, 2);

    // ---- kNN(x2) + edge block 3 -> x3 ----
    ne_cvt<<<(int)((B64N + 255) / 256), 256, 0, stream>>>(x2f, xh, (int)B64N);
    ne_sqnorm<<<(int)(BN / 256), 256, 0, stream>>>(xh, xx);
    ne_knn_feat<<<dim3(Nn / 16, Bb), 256, 0, stream>>>(xh, xx, idxf);
    ne_big_conv<<<dim3(Nn / 16, Bb), 256, 0, stream>>>(xh, idxf, conv3_w, gn3_w, gn3_b,
                                                       s3, inv_ck, x3f, 1);
    ne_big_conv<<<dim3(Nn / 16, Bb), 256, 0, stream>>>(xh, idxf, conv3_w, gn3_w, gn3_b,
                                                       s3, inv_ck, x3f, 2);

    // ---- fm + head ----
    ne_cvt<<<(256 * 192 + 255) / 256, 256, 0, stream>>>(fm_w, fmwh, 256 * 192);
    ne_fm<<<dim3(Nn / 64, Bb), 256, 0, stream>>>(x1f, x2f, x3f, fmwh, fm_b, yfm, s4);
    ne_head<<<(int)(BN / 256), 256, 0, stream>>>(yfm, s4, gn4_w, gn4_b, head_w, head_b,
                                                 (float*)d_out);
}
